// MultiHeadAttention_14001593385059
// MI455X (gfx1250) — compile-verified, matrix operand order checked by reading, not run on hardware
//
#include <hip/hip_runtime.h>
#include <hip/hip_fp16.h>

// ---------------------------------------------------------------------------
// MI455X (gfx1250) MHA:
//   cvt f32->f16 (activations) + cvt/transpose f32->f16 (weights)
//   -> Q/K/V projections: TDM double-buffered f16 WMMA GEMM
//   -> flash attention:   TDM double-buffered LDS + WMMA (S^T formulation)
//   -> output projection: same TDM WMMA GEMM
// wave32, V_WMMA_F32_16X16X32_F16 everywhere, tensor_load_to_lds staging.
// ---------------------------------------------------------------------------

typedef __attribute__((ext_vector_type(16))) _Float16 v16h;
typedef __attribute__((ext_vector_type(8)))  _Float16 v8h;
typedef __attribute__((ext_vector_type(2)))  _Float16 v2h;
typedef __attribute__((ext_vector_type(8)))  float    v8f;
typedef __attribute__((ext_vector_type(4)))  unsigned int u32x4;
typedef __attribute__((ext_vector_type(8)))  int          i32x8;
typedef __attribute__((ext_vector_type(4)))  int          i32x4;

#define BS    2
#define SEQ   2048
#define NH    16
#define HDIM  64
#define FEAT  1024          // QS = KS = VS = HD = 1024
#define MTOT  (BS * SEQ)    // 4096 rows for all projections

union V16 { v16h v; v8h h[2]; };

static __device__ __forceinline__ v8f wmma_f16(const V16& a, const V16& b, v8f c) {
  // D = A(16x32 f16) x B(32x16 f16) + C(16x16 f32)
  return __builtin_amdgcn_wmma_f32_16x16x32_f16(false, a.v, false, b.v,
                                                (short)0, c, false, false);
}

// ---------------------------------------------------------------------------
// Tensor Data Mover: 2D tile (f16 elements) global -> LDS, with LDS row
// padding so fragment ds_load_b128s are bank-conflict-free and 16B aligned.
// D# layout per CDNA5 ISA.
// ---------------------------------------------------------------------------
static __device__ __forceinline__ void tdm_load_2d_f16(
    const _Float16* gptr, unsigned int lds_off,
    unsigned int tensor_d0, unsigned int tensor_d1,
    unsigned int tile_d0,   unsigned int tile_d1,
    unsigned long long d0_stride,          // row stride in elements
    unsigned int pad_interval_code, unsigned int pad_amount_code)
{
  unsigned long long ga = (unsigned long long)(uintptr_t)gptr;
  u32x4 g0;
  g0[0] = 1u;                                        // count=1, user descriptor
  g0[1] = lds_off;                                   // lds_addr (bytes)
  g0[2] = (unsigned int)(ga & 0xffffffffu);          // global_addr[31:0]
  g0[3] = (unsigned int)((ga >> 32) & 0x01ffffffu)   // global_addr[56:32]
        | 0x80000000u;                               // type=2 ("image")
  i32x8 g1;
  unsigned int w0 = (1u << 16);                      // data_size = 2 bytes
  if (pad_amount_code | pad_interval_code)
    w0 |= (1u << 20) | (pad_interval_code << 22) | (pad_amount_code << 25);
  g1[0] = (int)w0;
  g1[1] = (int)(tensor_d0 << 16);                    // tensor_dim0[15:0]
  g1[2] = (int)((tensor_d0 >> 16) | (tensor_d1 << 16));
  g1[3] = (int)((tensor_d1 >> 16) | (tile_d0 << 16));
  g1[4] = (int)(tile_d1 & 0xffffu);                  // tile_dim1 (tile_dim2=0)
  g1[5] = (int)(unsigned int)(d0_stride & 0xffffffffu);
  g1[6] = (int)(unsigned int)((d0_stride >> 32) & 0xffffu);
  g1[7] = 0;
  i32x4 z4 = {0, 0, 0, 0};
#if __clang_major__ >= 23
  i32x8 z8 = {0, 0, 0, 0, 0, 0, 0, 0};
  __builtin_amdgcn_tensor_load_to_lds(g0, g1, z4, z4, z8, 0);
#else
  __builtin_amdgcn_tensor_load_to_lds(g0, g1, z4, z4, 0);
#endif
}

// ---------------------------------------------------------------------------
// One-pass converts (bandwidth-trivial: ~64MB total @ 23.3 TB/s)
// ---------------------------------------------------------------------------
__global__ __launch_bounds__(256) void cvt_f32_to_f16(
    const float* __restrict__ x, _Float16* __restrict__ y)
{
  const size_t i = ((size_t)blockIdx.x * 256 + threadIdx.x) * 8;
  float4 a = ((const float4*)(x + i))[0];
  float4 b = ((const float4*)(x + i))[1];
  v8h o = { (_Float16)a.x,(_Float16)a.y,(_Float16)a.z,(_Float16)a.w,
            (_Float16)b.x,(_Float16)b.y,(_Float16)b.z,(_Float16)b.w };
  *(v8h*)(y + i) = o;
}

// W [K][N] f32 -> Wt [N][K] f16, 32x32 LDS-tiled transpose
__global__ __launch_bounds__(256) void cvt_transpose(
    const float* __restrict__ W, _Float16* __restrict__ Wt, int K, int N)
{
  __shared__ _Float16 tile[32][33];
  const int tx = threadIdx.x & 31, ty = threadIdx.x >> 5;
  const int k0 = blockIdx.y * 32, n0 = blockIdx.x * 32;
  #pragma unroll
  for (int i = 0; i < 32; i += 8)
    tile[ty + i][tx] = (_Float16)W[(size_t)(k0 + ty + i) * N + n0 + tx];
  __syncthreads();
  #pragma unroll
  for (int i = 0; i < 32; i += 8)
    Wt[(size_t)(n0 + ty + i) * K + k0 + tx] = tile[tx][ty + i];
}

// ---------------------------------------------------------------------------
// All-f16 GEMM: C = (A(MxK) * Wt(NxK)^T + bias) * outScale.
// Block tile 128(M) x 64(N), k-step 32, 8 waves, each 32x32 (2x2 WMMA tiles).
// TDM double-buffers the A tile (128x32) and Wt tile (64x32) in LDS; rows are
// 16 dwords, padded by 4 -> 20-dword stride (conflict-free, 16B aligned).
// MODE 0: f32 out row-major [M][N]            (output projection -> d_out)
// MODE 1: f16 out head-major [B][H][L][64]    (Q pre-scaled by 1/8, K)
// MODE 2: f16 out head-major-T [B][H][64][L]  (V, pre-transposed for O=V^T.P^T)
// ---------------------------------------------------------------------------
template<int MODE>
__global__ __launch_bounds__(256) void gemm_f16(
    const _Float16* __restrict__ A, const _Float16* __restrict__ Wt,
    const float* __restrict__ bias, void* __restrict__ outp,
    int M, int K, int N, float outScale)
{
  constexpr int LS = 40;                   // padded LDS row stride (halves)
  __shared__ _Float16 Aa[2][128 * LS];
  __shared__ _Float16 Bb[2][ 64 * LS];

  const int tid  = threadIdx.x;
  const int wid  = tid >> 5;
  const int lane = tid & 31;
  const int hs   = lane >> 4;
  const int lr   = lane & 15;
  const int wrow = wid >> 1;               // 4 wave-rows * 32 M
  const int wcol = wid & 1;                // 2 wave-cols * 32 N
  const int m0   = blockIdx.x * 128;
  const int n0   = blockIdx.y * 64;

  const _Float16* Ablk = A  + (size_t)m0 * K;
  const _Float16* Bblk = Wt + (size_t)n0 * K;

  if (wid == 0) {                          // prefetch k-step 0 into buffer 0
    tdm_load_2d_f16(Ablk, (unsigned int)(uintptr_t)&Aa[0][0],
                    32, 128, 32, 128, (unsigned long long)K, 3, 3);
    tdm_load_2d_f16(Bblk, (unsigned int)(uintptr_t)&Bb[0][0],
                    32,  64, 32,  64, (unsigned long long)K, 3, 3);
  }

  v8f acc[2][2] = {};

  for (int k0 = 0; k0 < K; k0 += 32) {
    const int buf = (k0 >> 5) & 1;
    if (wid == 0) __builtin_amdgcn_s_wait_tensorcnt(0);
    __syncthreads();                       // publish buffer `buf`
    if (wid == 0 && k0 + 32 < K) {         // prefetch next k-step into buf^1
      tdm_load_2d_f16(Ablk + k0 + 32, (unsigned int)(uintptr_t)&Aa[buf ^ 1][0],
                      32, 128, 32, 128, (unsigned long long)K, 3, 3);
      tdm_load_2d_f16(Bblk + k0 + 32, (unsigned int)(uintptr_t)&Bb[buf ^ 1][0],
                      32,  64, 32,  64, (unsigned long long)K, 3, 3);
    }

    V16 af[2], bf[2];
    #pragma unroll
    for (int i = 0; i < 2; ++i) {
      const _Float16* ap = &Aa[buf][(wrow * 32 + i * 16 + lr) * LS + hs * 8];
      af[i].h[0] = *(const v8h*)ap;
      af[i].h[1] = *(const v8h*)(ap + 16);
    }
    #pragma unroll
    for (int j = 0; j < 2; ++j) {
      const _Float16* bp = &Bb[buf][(wcol * 32 + j * 16 + lr) * LS + hs * 8];
      bf[j].h[0] = *(const v8h*)bp;
      bf[j].h[1] = *(const v8h*)(bp + 16);
    }
    #pragma unroll
    for (int i = 0; i < 2; ++i)
      #pragma unroll
      for (int j = 0; j < 2; ++j)
        acc[i][j] = wmma_f16(af[i], bf[j], acc[i][j]);
  }

  // Epilogue. C-frag layout: VGPR r -> row (r + 8*hs), lane%16 -> col.
  #pragma unroll
  for (int i = 0; i < 2; ++i) {
    #pragma unroll
    for (int j = 0; j < 2; ++j) {
      #pragma unroll
      for (int r = 0; r < 8; ++r) {
        const int gm = m0 + wrow * 32 + i * 16 + r + 8 * hs;
        const int gn = n0 + wcol * 32 + j * 16 + lr;
        float val = (acc[i][j][r] + bias[gn]) * outScale;
        if constexpr (MODE == 0) {
          ((float*)outp)[(size_t)gm * N + gn] = val;
        } else {
          const int b = gm >> 11, q = gm & (SEQ - 1);
          const int h = gn >> 6,  d = gn & (HDIM - 1);
          if constexpr (MODE == 1)
            ((_Float16*)outp)[(((size_t)(b * NH + h)) * SEQ + q) * HDIM + d] =
                (_Float16)val;
          else
            ((_Float16*)outp)[(((size_t)(b * NH + h)) * HDIM + d) * SEQ + q] =
                (_Float16)val;
        }
      }
    }
  }
}

// ---------------------------------------------------------------------------
// Flash attention, S^T formulation. One wave = 16 queries (lane dimension),
// 32 keys per step:
//   S^T = K . Q^T    (4 WMMAs; keys land in the VGPR index of the C-frag)
//   softmax over keys = per-lane reduce + ONE shfl_xor(16)
//   P^T B-fragment   = in-lane f32->f16 convert of the S^T C-frag (no LDS)
//   O^T += V^T . P^T (4 WMMAs)
// K/V tiles staged in double-buffered LDS by the TDM.
//   K tile 32x64: rows 32 dw -> pad 4 dw  (stride 36 dw / 72 halves)
//   V tile 64x32: rows 16 dw -> pad 4 dw  (stride 20 dw / 40 halves)
// ---------------------------------------------------------------------------
__global__ __launch_bounds__(256) void attn_fused(
    const _Float16* __restrict__ Qh, const _Float16* __restrict__ Kh,
    const _Float16* __restrict__ Vt, _Float16* __restrict__ heads)
{
  __shared__ _Float16 Kl[2][32 * 72];
  __shared__ _Float16 Vl[2][64 * 40];

  const int tid  = threadIdx.x;
  const int wid  = tid >> 5;
  const int lane = tid & 31;
  const int hs   = lane >> 4;
  const int lr   = lane & 15;

  constexpr int QCH = SEQ / 128;
  const int qc = blockIdx.x % QCH;
  const int h  = (blockIdx.x / QCH) % NH;
  const int b  =  blockIdx.x / (QCH * NH);
  const int q0 = qc * 128 + wid * 16;

  const _Float16* Qb = Qh + ((size_t)(b * NH + h)) * SEQ * HDIM;
  const _Float16* Kb = Kh + ((size_t)(b * NH + h)) * SEQ * HDIM;
  const _Float16* Vb = Vt + ((size_t)(b * NH + h)) * HDIM * SEQ;

  // Q^T B-fragments: lane = query column, halves = d chunk (hs*8..+7, +16)
  V16 qb[2];
  #pragma unroll
  for (int c = 0; c < 2; ++c) {
    const _Float16* qp = Qb + (size_t)(q0 + lr) * HDIM + c * 32 + hs * 8;
    qb[c].h[0] = *(const v8h*)qp;
    qb[c].h[1] = *(const v8h*)(qp + 16);
  }

  if (wid == 0) {                          // prefetch first K/V pair -> buf 0
    tdm_load_2d_f16(Kb, (unsigned int)(uintptr_t)&Kl[0][0],
                    HDIM, 32, HDIM, 32, HDIM, 4, 3);
    tdm_load_2d_f16(Vb, (unsigned int)(uintptr_t)&Vl[0][0],
                    32, HDIM, 32, HDIM, SEQ, 3, 3);
  }

  float mcur = -3.0e38f, lcur = 0.f;       // per-lane: stats of query q0+lr
  v8f o[4] = {};                           // O^T d-tiles

  for (int kb = 0; kb < SEQ; kb += 32) {
    const int buf = (kb >> 5) & 1;
    if (wid == 0) __builtin_amdgcn_s_wait_tensorcnt(0);
    __syncthreads();                       // publish buffer `buf`
    if (wid == 0 && kb + 32 < SEQ) {       // prefetch next pair into buf^1
      tdm_load_2d_f16(Kb + (size_t)(kb + 32) * HDIM,
                      (unsigned int)(uintptr_t)&Kl[buf ^ 1][0],
                      HDIM, 32, HDIM, 32, HDIM, 4, 3);
      tdm_load_2d_f16(Vb + (kb + 32),
                      (unsigned int)(uintptr_t)&Vl[buf ^ 1][0],
                      32, HDIM, 32, HDIM, SEQ, 3, 3);
    }

    // S^T tiles: t=0 -> keys kb..kb+15, t=1 -> keys kb+16..kb+31
    v8f st[2] = { {}, {} };
    #pragma unroll
    for (int t = 0; t < 2; ++t) {
      #pragma unroll
      for (int c = 0; c < 2; ++c) {
        const _Float16* kp = &Kl[buf][(t * 16 + lr) * 72 + c * 32 + hs * 8];
        V16 ak;
        ak.h[0] = *(const v8h*)kp;
        ak.h[1] = *(const v8h*)(kp + 16);
        st[t] = wmma_f16(ak, qb[c], st[t]);
      }
    }

    // Online softmax for query lr: keys live in (t, r, hs). One shuffle only.
    float mx = st[0][0];
    #pragma unroll
    for (int r = 1; r < 8; ++r) mx = fmaxf(mx, st[0][r]);
    #pragma unroll
    for (int r = 0; r < 8; ++r) mx = fmaxf(mx, st[1][r]);
    mx = fmaxf(mx, __shfl_xor(mx, 16, 32));
    const float mnew = fmaxf(mcur, mx);
    const float corr = __expf(mcur - mnew);
    mcur = mnew;

    // P^T B-fragment: half j of chunk t == key kb + t*16 + hs*8 + j, which is
    // exactly st[t][j] in this lane. Pure in-lane convert.
    V16 pb;
    float ps = 0.f;
    #pragma unroll
    for (int r = 0; r < 8; ++r) {
      float p0 = __expf(st[0][r] - mnew);
      float p1 = __expf(st[1][r] - mnew);
      ps += p0 + p1;
      pb.h[0][r] = (_Float16)p0;
      pb.h[1][r] = (_Float16)p1;
    }
    ps += __shfl_xor(ps, 16, 32);
    lcur = lcur * corr + ps;
    #pragma unroll
    for (int t = 0; t < 4; ++t)
      #pragma unroll
      for (int r = 0; r < 8; ++r) o[t][r] *= corr;

    // O^T += V^T . P^T : A = V^T d-tile rows from padded LDS
    #pragma unroll
    for (int t = 0; t < 4; ++t) {
      const _Float16* vp = &Vl[buf][(t * 16 + lr) * 40 + hs * 8];
      V16 av;
      av.h[0] = *(const v8h*)vp;
      av.h[1] = *(const v8h*)(vp + 16);
      o[t] = wmma_f16(av, pb, o[t]);
    }
  }

  // heads [B][SEQ][NH*HDIM] f16, packed pair stores (d, d+1 adjacent in r).
  const float inv = 1.0f / lcur;
  _Float16* hp = heads + ((size_t)b * SEQ + q0 + lr) * (NH * HDIM) + h * HDIM;
  #pragma unroll
  for (int t = 0; t < 4; ++t)
    #pragma unroll
    for (int j = 0; j < 4; ++j) {
      v2h pk = { (_Float16)(o[t][2 * j]     * inv),
                 (_Float16)(o[t][2 * j + 1] * inv) };
      *(v2h*)&hp[t * 16 + 2 * j + 8 * hs] = pk;
    }
}

// ---------------------------------------------------------------------------
extern "C" void kernel_launch(void* const* d_in, const int* in_sizes, int n_in,
                              void* d_out, int out_size, void* d_ws, size_t ws_size,
                              hipStream_t stream) {
  (void)in_sizes; (void)n_in; (void)out_size; (void)ws_size;
  const float* query = (const float*)d_in[0];
  const float* key   = (const float*)d_in[1];
  const float* value = (const float*)d_in[2];
  const float* Wq_w  = (const float*)d_in[3];
  const float* Wq_b  = (const float*)d_in[4];
  const float* Wk_w  = (const float*)d_in[5];
  const float* Wk_b  = (const float*)d_in[6];
  const float* Wv_w  = (const float*)d_in[7];
  const float* Wv_b  = (const float*)d_in[8];
  const float* Wo_w  = (const float*)d_in[9];
  const float* Wo_b  = (const float*)d_in[10];

  // workspace (f16 elements): qf|kf|vf (4M each) | Wqt|Wkt|Wvt|Wot (1M each)
  //                           | Qh|Kh|Vt (4M each);  heads aliases qf (dead).
  const size_t act = (size_t)MTOT * FEAT;      // 4M
  const size_t wgt = (size_t)FEAT * FEAT;      // 1M
  _Float16* qf  = (_Float16*)d_ws;
  _Float16* kf  = qf  + act;
  _Float16* vf  = kf  + act;
  _Float16* Wqt = vf  + act;
  _Float16* Wkt = Wqt + wgt;
  _Float16* Wvt = Wkt + wgt;
  _Float16* Wot = Wvt + wgt;
  _Float16* Qh  = Wot + wgt;
  _Float16* Kh  = Qh  + act;
  _Float16* Vt  = Kh  + act;
  _Float16* heads = qf;                        // reuse: qf dead after Q proj

  dim3 blk(256);
  // converts
  cvt_f32_to_f16<<<dim3(act / (256 * 8)), blk, 0, stream>>>(query, qf);
  cvt_f32_to_f16<<<dim3(act / (256 * 8)), blk, 0, stream>>>(key,   kf);
  cvt_f32_to_f16<<<dim3(act / (256 * 8)), blk, 0, stream>>>(value, vf);
  dim3 gtr(FEAT / 32, FEAT / 32);
  cvt_transpose<<<gtr, blk, 0, stream>>>(Wq_w, Wqt, FEAT, FEAT);
  cvt_transpose<<<gtr, blk, 0, stream>>>(Wk_w, Wkt, FEAT, FEAT);
  cvt_transpose<<<gtr, blk, 0, stream>>>(Wv_w, Wvt, FEAT, FEAT);
  cvt_transpose<<<gtr, blk, 0, stream>>>(Wo_w, Wot, FEAT, FEAT);
  // projections (Q pre-scaled by 1/sqrt(D))
  dim3 gproj(MTOT / 128, FEAT / 64);
  gemm_f16<1><<<gproj, blk, 0, stream>>>(qf, Wqt, Wq_b, (void*)Qh,
                                         MTOT, FEAT, FEAT, 0.125f);
  gemm_f16<1><<<gproj, blk, 0, stream>>>(kf, Wkt, Wk_b, (void*)Kh,
                                         MTOT, FEAT, FEAT, 1.0f);
  gemm_f16<2><<<gproj, blk, 0, stream>>>(vf, Wvt, Wv_b, (void*)Vt,
                                         MTOT, FEAT, FEAT, 1.0f);
  // attention
  attn_fused<<<dim3(BS * NH * (SEQ / 128)), blk, 0, stream>>>(Qh, Kh, Vt, heads);
  // output projection
  gemm_f16<0><<<gproj, blk, 0, stream>>>(heads, Wot, Wo_b, d_out,
                                         MTOT, FEAT, FEAT, 1.0f);
}
